// KnowledgeFuseBlock_23974507446297
// MI455X (gfx1250) — matrix operand order repaired and checked by reading, not compiled, not run on hardware
//
#include <hip/hip_runtime.h>
#include <hip/hip_bf16.h>

// ---------------------------------------------------------------------------
// Types
// ---------------------------------------------------------------------------
typedef unsigned short u16;
typedef __attribute__((ext_vector_type(16))) __bf16          v16bf;
typedef __attribute__((ext_vector_type(16))) unsigned short  v16u;
typedef __attribute__((ext_vector_type(8)))  unsigned short  u16x8;
typedef __attribute__((ext_vector_type(8)))  float           v8f;

__device__ __forceinline__ u16 f2bf(float f) {
    unsigned u = __builtin_bit_cast(unsigned, f);
    unsigned r = u + 0x7FFFu + ((u >> 16) & 1u);   // round-to-nearest-even
    return (u16)(r >> 16);
}
__device__ __forceinline__ float leaky(float x) { return x > 0.f ? x : 0.01f * x; }

// CDNA5 async memory->LDS copy (16B per lane), tracked by ASYNCcnt.
__device__ __forceinline__ void async_copy_b128(unsigned lds_byte_off, const void* gptr) {
    asm volatile("global_load_async_to_lds_b128 %0, %1, off"
                 :: "v"(lds_byte_off), "v"((unsigned long long)(uintptr_t)gptr)
                 : "memory");
}
__device__ __forceinline__ void wait_async0() {
    asm volatile("s_wait_asynccnt 0x0" ::: "memory");
}
__device__ __forceinline__ unsigned lds_off(const void* p) {
    // addrspace(3) offset == low 32 bits of the generic pointer
    return (unsigned)(uintptr_t)p;
}

// ---------------------------------------------------------------------------
// Epilogue descriptor
// ---------------------------------------------------------------------------
#define EPI_NONE       0
#define EPI_LEAKY_BN   1   // b,g,be,m,v indexed by n
#define EPI_LEAKY_BIAS 2   // b indexed by n
#define EPI_TANH_BIAS  3   // b indexed by n
#define EPI_RESIDUAL   4   // res[m*ldres+n] f32
#define EPI_DIST       5   // sqrt(max(row[m] + b[n] - 2*acc, 0))

struct EpiParams {
    int type;
    const float* b;
    const float* g;
    const float* be;
    const float* m;
    const float* v;
    const float* row;
    const float* res;
    int ldres;
    int out_bf16;
};

__device__ __forceinline__ float apply_epi(float a, int m, int n, const EpiParams& ep) {
    switch (ep.type) {
    case EPI_LEAKY_BN: {
        float x = leaky(a + ep.b[n]);
        float s = ep.g[n] * rsqrtf(ep.v[n] + 1e-5f);
        return (x - ep.m[n]) * s + ep.be[n];
    }
    case EPI_LEAKY_BIAS: return leaky(a + ep.b[n]);
    case EPI_TANH_BIAS:  return tanhf(a + ep.b[n]);
    case EPI_RESIDUAL:   return a + ep.res[(size_t)m * ep.ldres + n];
    case EPI_DIST: {
        float d2 = ep.row[m] + ep.b[n] - 2.f * a;
        return sqrtf(fmaxf(d2, 0.f));
    }
    default: return a;
    }
}

// ---------------------------------------------------------------------------
// WMMA GEMM: C[M,N] = epi(A[M,K](bf16) * B(bf16)), B row-major [K,N] (transB=0)
// or [N,K] (transB=1).  Block tile 128x64, BK=32, 256 threads = 8 wave32.
// Wave grid 4(m) x 2(n); each wave owns 32x32 -> 2 A-frags x 2 B-frags ->
// 4 x v_wmma_f32_16x16x32_bf16 per K step.
// Requires: M % 128 == 0, N % 64 == 0, K % 32 == 0, 16B-aligned A/B.
// ---------------------------------------------------------------------------
#define BM 128
#define BN 64
#define BK 32
#define LDSP 40   // row stride in halves (80 B, 16B aligned)

__global__ __launch_bounds__(256)
void gemm_wmma_bf16(const u16* __restrict__ A, const u16* __restrict__ B,
                    void* __restrict__ C, int M, int N, int K,
                    int transB, EpiParams ep) {
    __shared__ __align__(16) u16 As[BM][LDSP];
    __shared__ __align__(16) u16 Bs[BN][LDSP];

    const int tid  = threadIdx.x;
    const int wave = tid >> 5;
    const int lane = tid & 31;
    const int block_m = blockIdx.y * BM;
    const int block_n = blockIdx.x * BN;
    const int wm = (wave & 3) * 32;   // wave M offset within block
    const int wn = (wave >> 2) * 32;  // wave N offset within block

    const unsigned AsB = lds_off(&As[0][0]);
    const unsigned BsB = lds_off(&Bs[0][0]);

    v8f acc00 = {}, acc01 = {}, acc10 = {}, acc11 = {};

    for (int k0 = 0; k0 < K; k0 += BK) {
        // ---- stage A tile [128 x 32] via async b128 copies (2 per thread) ----
#pragma unroll
        for (int i = 0; i < 2; i++) {
            int tt = tid + i * 256;
            int r = tt >> 2, sg = tt & 3;
            async_copy_b128(AsB + (unsigned)(r * (LDSP * 2) + sg * 16),
                            &A[(size_t)(block_m + r) * K + (k0 + sg * 8)]);
        }
        // ---- stage B tile into Bs[n][k] ----
        if (transB) {  // B is [N,K]: contiguous along k -> async b128
            int r = tid >> 2, sg = tid & 3;
            async_copy_b128(BsB + (unsigned)(r * (LDSP * 2) + sg * 16),
                            &B[(size_t)(block_n + r) * K + (k0 + sg * 8)]);
        } else {       // B is [K,N]: vector load along n, register transpose
            int kk = tid >> 3, nseg = tid & 7;
            u16x8 bv = *(const u16x8*)&B[(size_t)(k0 + kk) * N + block_n + nseg * 8];
#pragma unroll
            for (int i = 0; i < 8; i++) Bs[nseg * 8 + i][kk] = bv[i];
        }
        // prefetch next A tile into L2 (global_prefetch_b8 path)
        if (k0 + BK < K)
            __builtin_prefetch(&A[(size_t)(block_m + (tid >> 1)) * K + (k0 + BK) + (tid & 1) * 16], 0, 0);

        wait_async0();
        __syncthreads();

        // ---- fragments per CDNA5 WMMA VGPR layouts ----
        // A: lane holds row (lane&15); lanes 0-15 K{0-7,16-23}, lanes 16-31 K{8-15,24-31}
        v16u au0, au1, bu0, bu1;
        const int ar0 = wm + (lane & 15);
        const int akb = (lane >> 4) ? 8 : 0;
#pragma unroll
        for (int vv = 0; vv < 8; vv++) {
            int kk = ((vv >= 4) ? 16 : 0) + akb + (vv & 3) * 2;
            au0[2 * vv]     = As[ar0][kk];
            au0[2 * vv + 1] = As[ar0][kk + 1];
            au1[2 * vv]     = As[ar0 + 16][kk];
            au1[2 * vv + 1] = As[ar0 + 16][kk + 1];
        }
        // B: lane holds col (lane&15); lanes 0-15 K{0-7,16-23}, lanes 16-31 K{8-15,24-31}
        const int bc0 = wn + (lane & 15);
        const int bkb = (lane >> 4) ? 8 : 0;
#pragma unroll
        for (int vv = 0; vv < 8; vv++) {
            int kk = ((vv >= 4) ? 16 : 0) + bkb + (vv & 3) * 2;
            bu0[2 * vv]     = Bs[bc0][kk];
            bu0[2 * vv + 1] = Bs[bc0][kk + 1];
            bu1[2 * vv]     = Bs[bc0 + 16][kk];
            bu1[2 * vv + 1] = Bs[bc0 + 16][kk + 1];
        }
        v16bf a0 = __builtin_bit_cast(v16bf, au0);
        v16bf a1 = __builtin_bit_cast(v16bf, au1);
        v16bf b0 = __builtin_bit_cast(v16bf, bu0);
        v16bf b1 = __builtin_bit_cast(v16bf, bu1);

        acc00 = __builtin_amdgcn_wmma_f32_16x16x32_bf16(false, a0, false, b0, (short)0, acc00, false, false);
        acc01 = __builtin_amdgcn_wmma_f32_16x16x32_bf16(false, a0, false, b1, (short)0, acc01, false, false);
        acc10 = __builtin_amdgcn_wmma_f32_16x16x32_bf16(false, a1, false, b0, (short)0, acc10, false, false);
        acc11 = __builtin_amdgcn_wmma_f32_16x16x32_bf16(false, a1, false, b1, (short)0, acc11, false, false);
        __syncthreads();
    }

    // ---- store; C/D layout: M = v + 8*(lane>=16), N = lane&15 ----
    const int lnn = lane & 15;
    const int mb  = (lane >> 4) * 8;
    auto store_tile = [&](const v8f& acc, int m0, int n0) {
#pragma unroll
        for (int vv = 0; vv < 8; vv++) {
            int m = block_m + m0 + mb + vv;
            int n = block_n + n0 + lnn;
            float val = apply_epi(acc[vv], m, n, ep);
            if (ep.out_bf16) ((u16*)C)[(size_t)m * N + n] = f2bf(val);
            else             ((float*)C)[(size_t)m * N + n] = val;
        }
    };
    store_tile(acc00, wm,      wn);
    store_tile(acc01, wm,      wn + 16);
    store_tile(acc10, wm + 16, wn);
    store_tile(acc11, wm + 16, wn + 16);
}

// ---------------------------------------------------------------------------
// Helper kernels
// ---------------------------------------------------------------------------
__global__ void f32_to_bf16_k(const float* __restrict__ s, u16* __restrict__ d, size_t n) {
    size_t i = (size_t)blockIdx.x * 256 + threadIdx.x;
    if (i < n) d[i] = f2bf(s[i]);
}

// dst[C x R] bf16 = transpose(src[R x C] f32)
__global__ void transpose_bf16_k(const float* __restrict__ src, u16* __restrict__ dst,
                                 int R, int C) {
    __shared__ float tile[32][33];
    int bx = blockIdx.x * 32, by = blockIdx.y * 32;
    int tx = threadIdx.x, ty = threadIdx.y;
#pragma unroll
    for (int j = 0; j < 32; j += 8) {
        int r = by + ty + j, c = bx + tx;
        if (r < R && c < C) tile[ty + j][tx] = src[(size_t)r * C + c];
    }
    __syncthreads();
#pragma unroll
    for (int j = 0; j < 32; j += 8) {
        int c = bx + ty + j, r = by + tx;
        if (c < C && r < R) dst[(size_t)c * R + r] = f2bf(tile[tx][ty + j]);
    }
}

// out[row] = sum_k X[row,k]^2 ; one wave32 per row
__global__ void rownorm2_k(const float* __restrict__ X, float* __restrict__ out,
                           int M, int Kd) {
    int row  = blockIdx.x * 8 + (threadIdx.x >> 5);
    int lane = threadIdx.x & 31;
    if (row >= M) return;
    float s = 0.f;
    for (int k = lane; k < Kd; k += 32) { float x = X[(size_t)row * Kd + k]; s += x * x; }
#pragma unroll
    for (int off = 16; off; off >>= 1) s += __shfl_xor(s, off, 32);
    if (lane == 0) out[row] = s;
}

// softmax over rows (axis=1 of [R,C]) per column; writes bf16
__global__ void col_softmax_bf16_k(const float* __restrict__ X, u16* __restrict__ out,
                                   int R, int C) {
    int col = blockIdx.x * 256 + threadIdx.x;
    if (col >= C) return;
    float mx = -3.4e38f;
    for (int i = 0; i < R; i++) mx = fmaxf(mx, X[(size_t)i * C + col]);
    float s = 0.f;
    for (int i = 0; i < R; i++) s += __expf(X[(size_t)i * C + col] - mx);
    float inv = 1.f / s;
    for (int i = 0; i < R; i++)
        out[(size_t)i * C + col] = f2bf(__expf(X[(size_t)i * C + col] - mx) * inv);
}

// wlog[j] = b4 + sum_h A3[h,j] * W4[h]  (A3: [256 x 2048])
__global__ void mask_logits_k(const float* __restrict__ A3, const float* __restrict__ W4,
                              const float* __restrict__ b4, float* __restrict__ wlog) {
    int j = blockIdx.x * 256 + threadIdx.x;
    if (j >= 2048) return;
    float s = b4[0];
    for (int h = 0; h < 256; h++) s += A3[(size_t)h * 2048 + j] * W4[h];
    wlog[j] = s;
}

// top-512 of wlog[2048] by rank counting; ties broken by lower index (matches top_k).
// (softmax is monotonic, so top-k of logits == top-k of softmax(w))
__global__ __launch_bounds__(1024)
void topk512_k(const float* __restrict__ wlog, int* __restrict__ idx) {
    __shared__ float wl[2048];
    int t = threadIdx.x;
    wl[t] = wlog[t]; wl[t + 1024] = wlog[t + 1024];
    __syncthreads();
#pragma unroll
    for (int e = 0; e < 2; e++) {
        int i = t + e * 1024;
        float wi = wl[i];
        int rank = 0;
        for (int j = 0; j < 2048; j++) {
            float wj = wl[j];
            rank += (wj > wi) || (wj == wi && j < i);
        }
        if (rank < 512) idx[rank] = i;
    }
}

// sel[i,s] = softmax_s(aff_skip[i, idx[s]]) written as bf16 into fused[:,512+s]
__global__ __launch_bounds__(512)
void gather_softmax_k(const float* __restrict__ aff_skip, const int* __restrict__ idx,
                      u16* __restrict__ fused) {
    __shared__ int   sidx[512];
    __shared__ float red[512];
    int i = blockIdx.x, t = threadIdx.x;
    sidx[t] = idx[t];
    __syncthreads();
    float v = aff_skip[(size_t)i * 2048 + sidx[t]];
    red[t] = v; __syncthreads();
    for (int off = 256; off; off >>= 1) { if (t < off) red[t] = fmaxf(red[t], red[t + off]); __syncthreads(); }
    float mx = red[0]; __syncthreads();
    float e = __expf(v - mx);
    red[t] = e; __syncthreads();
    for (int off = 256; off; off >>= 1) { if (t < off) red[t] += red[t + off]; __syncthreads(); }
    float inv = 1.f / red[0];
    fused[(size_t)i * 1024 + 512 + t] = f2bf(e * inv);
}

// fused[:, 0:512] = bf16(B)
__global__ void pack_B_fused_k(const float* __restrict__ B, u16* __restrict__ fused) {
    size_t i = (size_t)blockIdx.x * 256 + threadIdx.x;
    if (i >= (size_t)4096 * 512) return;
    size_t row = i >> 9, col = i & 511;
    fused[row * 1024 + col] = f2bf(B[i]);
}

// ---------------------------------------------------------------------------
// Host orchestration
// ---------------------------------------------------------------------------
extern "C" void kernel_launch(void* const* d_in, const int* in_sizes, int n_in,
                              void* d_out, int out_size, void* d_ws, size_t ws_size,
                              hipStream_t stream) {
    (void)in_sizes; (void)n_in; (void)out_size; (void)ws_size;
    const int NX = 4096, NBAG = 2048, Dd = 512, HC = 512, HK = 256;

    const float* Bin  = (const float*)d_in[0];
    const float* Kbag = (const float*)d_in[1];
    auto P = [&](int i) { return (const float*)d_in[i]; };

    // ---- workspace arena (all 256B aligned -> 16B async alignment holds) ----
    size_t off = 0;
    auto alloc = [&](size_t bytes) -> void* {
        void* p = (char*)d_ws + off;
        off += (bytes + 255) & ~(size_t)255;
        return p;
    };
    u16* wc1  = (u16*)alloc((size_t)HC * Dd * 2);
    u16* wc2  = (u16*)alloc((size_t)HC * Dd * 2);
    u16* wc3  = (u16*)alloc((size_t)Dd * Dd * 2);
    u16* w2c1 = (u16*)alloc((size_t)HC * Dd * 2);
    u16* w2c2 = (u16*)alloc((size_t)HC * Dd * 2);
    u16* w2c3 = (u16*)alloc((size_t)Dd * Dd * 2);
    u16* w3c1 = (u16*)alloc((size_t)HC * NBAG * 2);
    u16* w3c2 = (u16*)alloc((size_t)HC * NBAG * 2);
    u16* w3c3 = (u16*)alloc((size_t)NBAG * NBAG * 2);
    u16* wm1  = (u16*)alloc((size_t)HK * NBAG * 2);
    u16* wm2  = (u16*)alloc((size_t)HK * NBAG * 2);
    u16* wm3  = (u16*)alloc((size_t)NBAG * HK * 2);
    u16* wemb = (u16*)alloc((size_t)Dd * 1024 * 2);
    u16*   Bbf    = (u16*)alloc((size_t)NX * Dd * 2);
    u16*   Kbf    = (u16*)alloc((size_t)NBAG * Dd * 2);
    u16*   Y1     = (u16*)alloc((size_t)NX * HC * 2);   // reused by skip/skip2/skip3
    u16*   Y2     = (u16*)alloc((size_t)NX * HC * 2);
    u16*   Y3     = (u16*)alloc((size_t)NX * Dd * 2);
    float* feat   = (float*)alloc((size_t)NX * NX * 4); // reused 3x
    u16*   featbf = (u16*)alloc((size_t)NX * NX * 2);
    float* Xs     = (float*)alloc((size_t)NX * Dd * 4);
    u16*   Xsbf   = (u16*)alloc((size_t)NX * Dd * 2);
    float* xs2    = (float*)alloc((size_t)NX * 4);
    float* Ks     = (float*)alloc((size_t)NBAG * Dd * 4);
    u16*   Ksbf   = (u16*)alloc((size_t)NBAG * Dd * 2);
    float* ks2    = (float*)alloc((size_t)NBAG * 4);
    float* aff    = (float*)alloc((size_t)NX * NBAG * 4);
    u16*   affbf  = (u16*)alloc((size_t)NX * NBAG * 2);
    u16*   Z3     = (u16*)alloc((size_t)NX * NBAG * 2);
    float* affS   = (float*)alloc((size_t)NX * NBAG * 4);
    u16*   affSbf = (u16*)alloc((size_t)NX * NBAG * 2);
    float* M1     = (float*)alloc((size_t)NX * HK * 4);
    u16*   M1T    = (u16*)alloc((size_t)HK * NX * 2);
    u16*   M2bf   = (u16*)alloc((size_t)NX * HK * 2);
    float* Sf     = (float*)alloc((size_t)HK * HK * 4);
    u16*   Sbf    = (u16*)alloc((size_t)HK * HK * 2);
    float* A3     = (float*)alloc((size_t)HK * NBAG * 4);
    float* wlog   = (float*)alloc((size_t)NBAG * 4);
    int*   idxb   = (int*)alloc((size_t)512 * 4);
    u16*   fused  = (u16*)alloc((size_t)NX * 1024 * 2);

    auto cvt = [&](const float* s, u16* d, size_t n) {
        f32_to_bf16_k<<<(unsigned)((n + 255) / 256), 256, 0, stream>>>(s, d, n);
    };
    auto gemm = [&](const u16* A, const u16* Bm, void* C, int M, int N, int K,
                    int transB, EpiParams ep) {
        gemm_wmma_bf16<<<dim3(N / BN, M / BM), 256, 0, stream>>>(A, Bm, C, M, N, K, transB, ep);
    };
    auto epNone = [&](int obf) { EpiParams e{}; e.type = EPI_NONE; e.out_bf16 = obf; return e; };
    auto epBN = [&](int base, int obf) {
        EpiParams e{}; e.type = EPI_LEAKY_BN;
        e.b = P(base + 1); e.g = P(base + 2); e.be = P(base + 3); e.m = P(base + 4); e.v = P(base + 5);
        e.out_bf16 = obf; return e;
    };
    auto epBias = [&](int bi, int type, int obf) {
        EpiParams e{}; e.type = type; e.b = P(bi); e.out_bf16 = obf; return e;
    };
    auto epRes = [&](const float* r, int ld) {
        EpiParams e{}; e.type = EPI_RESIDUAL; e.res = r; e.ldres = ld; return e;
    };
    auto epDist = [&](const float* rv, const float* cv) {
        EpiParams e{}; e.type = EPI_DIST; e.row = rv; e.b = cv; return e;
    };

    // ---- weight + input conversion (f32 -> bf16) ----
    cvt(P(2),  wc1,  (size_t)HC * Dd);   cvt(P(8),  wc2,  (size_t)HC * Dd);
    cvt(P(14), wc3,  (size_t)Dd * Dd);
    cvt(P(20), w2c1, (size_t)HC * Dd);   cvt(P(26), w2c2, (size_t)HC * Dd);
    cvt(P(32), w2c3, (size_t)Dd * Dd);
    cvt(P(38), w3c1, (size_t)HC * NBAG); cvt(P(44), w3c2, (size_t)HC * NBAG);
    cvt(P(50), w3c3, (size_t)NBAG * NBAG);
    cvt(P(56), wm1,  (size_t)HK * NBAG); cvt(P(58), wm2, (size_t)HK * NBAG);
    cvt(P(60), wm3,  (size_t)NBAG * HK); cvt(P(64), wemb, (size_t)Dd * 1024);
    cvt(Bin,  Bbf, (size_t)NX * Dd);
    cvt(Kbag, Kbf, (size_t)NBAG * Dd);

    // ---- skip(B): n=4096, c=512 ----
    gemm(Bbf, wc1, Y1, NX, HC, Dd, 1, epBN(2, 1));
    gemm(Bbf, wc2, Y2, NX, HC, Dd, 1, epBN(8, 1));
    gemm(Bbf, wc3, Y3, NX, Dd, Dd, 1, epBN(14, 1));
    gemm(Y1, Y2, feat, NX, NX, HC, 1, epNone(0));            // feat = Y1 * Y2^T
    col_softmax_bf16_k<<<NX / 256, 256, 0, stream>>>(feat, featbf, NX, NX);
    gemm(featbf, Y3, Xs, NX, Dd, NX, 0, epRes(Bin, Dd));     // Xs = softmax(feat)*Y3 + B
    cvt(Xs, Xsbf, (size_t)NX * Dd);
    rownorm2_k<<<NX / 8, 256, 0, stream>>>(Xs, xs2, NX, Dd);

    // ---- skip2(knowledge_bag): n=2048 ----
    gemm(Kbf, w2c1, Y1, NBAG, HC, Dd, 1, epBN(20, 1));
    gemm(Kbf, w2c2, Y2, NBAG, HC, Dd, 1, epBN(26, 1));
    gemm(Kbf, w2c3, Y3, NBAG, Dd, Dd, 1, epBN(32, 1));
    gemm(Y1, Y2, feat, NBAG, NBAG, HC, 1, epNone(0));
    col_softmax_bf16_k<<<NBAG / 256, 256, 0, stream>>>(feat, featbf, NBAG, NBAG);
    gemm(featbf, Y3, Ks, NBAG, Dd, NBAG, 0, epRes(Kbag, Dd));
    cvt(Ks, Ksbf, (size_t)NBAG * Dd);
    rownorm2_k<<<NBAG / 8, 256, 0, stream>>>(Ks, ks2, NBAG, Dd);

    // ---- affinity: aff = sqrt(max(|x|^2 + |k|^2 - 2 x.k, 0)) ----
    gemm(Xsbf, Ksbf, aff, NX, NBAG, Dd, 1, epDist(xs2, ks2));
    cvt(aff, affbf, (size_t)NX * NBAG);

    // ---- skip3(aff): n=4096, c=2048 ----
    gemm(affbf, w3c1, Y1, NX, HC, NBAG, 1, epBN(38, 1));
    gemm(affbf, w3c2, Y2, NX, HC, NBAG, 1, epBN(44, 1));
    gemm(affbf, w3c3, Z3, NX, NBAG, NBAG, 1, epBN(50, 1));
    gemm(Y1, Y2, feat, NX, NX, HC, 1, epNone(0));
    col_softmax_bf16_k<<<NX / 256, 256, 0, stream>>>(feat, featbf, NX, NX);
    gemm(featbf, Z3, affS, NX, NBAG, NX, 0, epRes(aff, NBAG));
    cvt(affS, affSbf, (size_t)NX * NBAG);

    // ---- mask block ----
    gemm(affSbf, wm1, M1, NX, HK, NBAG, 1, epBias(57, EPI_LEAKY_BIAS, 0));
    transpose_bf16_k<<<dim3(HK / 32, NX / 32), dim3(32, 8), 0, stream>>>(M1, M1T, NX, HK);
    gemm(affSbf, wm2, M2bf, NX, HK, NBAG, 1, epBias(59, EPI_TANH_BIAS, 1));
    gemm(M1T, M2bf, Sf, HK, HK, NX, 0, epNone(0));           // X1 @ X2  [256x256]
    cvt(Sf, Sbf, (size_t)HK * HK);
    gemm(Sbf, wm3, A3, HK, NBAG, HK, 1, epBias(61, EPI_LEAKY_BIAS, 0));
    mask_logits_k<<<NBAG / 256, 256, 0, stream>>>(A3, P(62), P(63), wlog);
    topk512_k<<<1, 1024, 0, stream>>>(wlog, idxb);

    // ---- gather + softmax + fuse + final linear ----
    pack_B_fused_k<<<(NX * Dd) / 256, 256, 0, stream>>>(Bin, fused);
    gather_softmax_k<<<NX, 512, 0, stream>>>(affS, idxb, fused);
    gemm(fused, wemb, d_out, NX, Dd, 1024, 1, epBias(65, EPI_LEAKY_BIAS, 0));
}
